// SFSPolicy_59545426592265
// MI455X (gfx1250) — matrix operand order repaired and checked by reading, not compiled, not run on hardware
//
#include <hip/hip_runtime.h>
#include <hip/hip_bf16.h>

// ---------------------------------------------------------------------------
// CDNA5 (gfx1250) implementation: bf16 WMMA GEMMs (16x64 strip per wave,
// async-LDS double-buffered A staging) + scalar softmax/select kernels.
// ---------------------------------------------------------------------------

typedef __attribute__((ext_vector_type(16))) __bf16 v16bf;
typedef __attribute__((ext_vector_type(8)))  float  v8f;

#define BB    128
#define EE    128
#define HH    256
#define NMAXD 256
#define MAXAD 40
#define MM    100
#define AMAXD 10
#define DESCD 1024
#define WAVES_PER_BLOCK 8

// -------- gfx1250 async global->LDS path (probe-verified) ------------------
#if defined(__AMDGCN__) && \
    __has_builtin(__builtin_amdgcn_global_load_async_to_lds_b128) && \
    __has_builtin(__builtin_amdgcn_s_wait_asynccnt)
#define GEMM_ASYNC 1
typedef int v4i __attribute__((vector_size(16)));
typedef __attribute__((address_space(1))) v4i* as1_v4p;
typedef __attribute__((address_space(3))) v4i* as3_v4p;
__device__ __forceinline__ void async_copy_b128(const float* g, float* l)
{
  __builtin_amdgcn_global_load_async_to_lds_b128(
      (as1_v4p)(v4i*)(float*)(g), (as3_v4p)(v4i*)(l), 0, 0);
}
#else
#define GEMM_ASYNC 0
#endif

// ---------------------------------------------------------------------------
// Wave-level bf16 WMMA GEMM. Each wave computes a 16(M) x 64(N) strip with
// 4 independent accumulators (hides WMMA hazards, reuses the A fragment 4x).
// The K-loop is fully branchless: out-of-range rows/cols are CLAMPED on load
// (garbage flows into tiles that are never stored), masking happens only at
// the store. Tile coordinates are forced into SGPRs via readfirstlane so all
// control flow around the WMMAs is scalar and EXEC stays all-ones.
//   bLayout 0: B is (N x K) row-major  -> C = A @ B^T
//   bLayout 1: B is (K x N) row-major  -> C = A @ B
//   flags: 1 = relu, 2 = accumulate into C, 4 = store C transposed (n-major)
// ---------------------------------------------------------------------------
__global__ __launch_bounds__(256) void gemm_wmma_bf16(
    const float* __restrict__ A, const float* __restrict__ Bm,
    const float* __restrict__ bias, float* __restrict__ C,
    int M, int N, int K, int lda, int ldb, int ldc,
    long azs, long bzs, long czs,
    int bLayout, int flags, int tilesN4, int tilesTotal)
{
#if GEMM_ASYNC
  __shared__ float abuf[WAVES_PER_BLOCK * 2 * 16 * 32]; // 2KB x 2 buffers/wave
#endif
  const int wave = __builtin_amdgcn_readfirstlane(threadIdx.x >> 5); // SGPR
  const int lane = threadIdx.x & 31;
  const int tile = blockIdx.x * WAVES_PER_BLOCK + wave;              // SGPR
  if (tile >= tilesTotal) return;                                    // scalar

  const int tm  = tile / tilesN4;          // all SGPR
  const int tn4 = tile % tilesN4;
  const int m0  = tm * 16;
  const int n0  = tn4 * 64;
  const int z   = blockIdx.z;

  const float* Ab = A  + (long)z * azs;
  const float* Bb = Bm + (long)z * bzs;
  float*       Cb = C  + (long)z * czs;

  const int hf   = lane >> 4;
  const int l16  = lane & 15;
  const int mRow = m0 + l16;
  const int mC   = (mRow < M) ? mRow : (M - 1);       // clamped A row

  int nClamp[4];                                       // clamped B columns
#pragma unroll
  for (int t = 0; t < 4; ++t) {
    int nc = n0 + 16 * t + l16;
    nClamp[t] = (nc < N) ? nc : (N - 1);
  }

  v8f acc[4];
#pragma unroll
  for (int t = 0; t < 4; ++t) acc[t] = (v8f){0.f,0.f,0.f,0.f,0.f,0.f,0.f,0.f};

#if GEMM_ASYNC
  float* wbuf = &abuf[wave * 1024];
  // stage 16x32 fp32 A tile: 128 b128 chunks, 32 lanes x 4 issues
  auto issueA = [&](int kb, int bufIdx) {
    float* dst = wbuf + bufIdx * 512;
#pragma unroll
    for (int c4 = 0; c4 < 4; ++c4) {
      int c = lane + 32 * c4;
      int row = c >> 3, col = (c & 7) * 4;
      int r = m0 + row; if (r >= M) r = M - 1;
      async_copy_b128(Ab + (long)r * lda + kb + col, dst + row * 32 + col);
    }
  };
  issueA(0, 0);
  int buf = 0;
#endif

  for (int kb = 0; kb < K; kb += 32) {
    // ---- A fragment (16x32 MxK): lane holds row, elements 0..15 -> K=kb+16*hf+e
    v16bf af;
#if GEMM_ASYNC
    if (kb + 32 < K) { issueA(kb + 32, buf ^ 1); __builtin_amdgcn_s_wait_asynccnt(4); }
    else             { __builtin_amdgcn_s_wait_asynccnt(0); }
    {
      const float* ap = wbuf + buf * 512 + l16 * 32 + 16 * hf;   // LDS (ds_load)
#pragma unroll
      for (int e = 0; e < 8; ++e) af[e] = (__bf16)ap[e];
      ap += 8;
#pragma unroll
      for (int e = 0; e < 8; ++e) af[8 + e] = (__bf16)ap[e];
    }
    buf ^= 1;
#else
    {
      const float* ap = Ab + (long)mC * lda + kb + 16 * hf;
      __builtin_prefetch(ap + 32, 0, 1);
#pragma unroll
      for (int e = 0; e < 8; ++e) af[e] = (__bf16)ap[e];
      ap += 8;
#pragma unroll
      for (int e = 0; e < 8; ++e) af[8 + e] = (__bf16)ap[e];
    }
#endif

    // ---- B fragments (32x16 KxN per sub-tile): lane holds col, K=kb+16*hf+e
    v16bf bfr[4];
    if (bLayout == 0) {
#pragma unroll
      for (int t = 0; t < 4; ++t) {
        const float* bp = Bb + (long)nClamp[t] * ldb + kb + 16 * hf;
        if (t == 0) __builtin_prefetch(bp + 32, 0, 1);
#pragma unroll
        for (int e = 0; e < 16; ++e) bfr[t][e] = (__bf16)bp[e];
      }
    } else {
#pragma unroll
      for (int t = 0; t < 4; ++t) {
        const float* bp = Bb + (long)(kb + 16 * hf) * ldb + nClamp[t];
#pragma unroll
        for (int e = 0; e < 16; ++e) bfr[t][e] = (__bf16)bp[(long)e * ldb];
      }
    }

    // ---- 4 independent WMMAs, no guards: EXEC is all-ones here
#pragma unroll
    for (int t = 0; t < 4; ++t)
      acc[t] = __builtin_amdgcn_wmma_f32_16x16x32_bf16(
          false, af, false, bfr[t], (short)0, acc[t], false, false);
  }

  // ---- store (D layout: VGPR r -> M = m0 + r + 8*hf, N = lane&15) ----
#pragma unroll
  for (int t = 0; t < 4; ++t) {
    const int nn  = n0 + 16 * t + l16;
    const bool nOk = nn < N;
    const float bv = (bias != nullptr && nOk) ? bias[nn] : 0.0f;
#pragma unroll
    for (int r = 0; r < 8; ++r) {
      const int mo = m0 + r + 8 * hf;
      if (mo >= M || !nOk) continue;
      const long addr = (flags & 4) ? ((long)nn * ldc + mo)
                                    : ((long)mo * ldc + nn);
      float v = acc[t][r] + bv;
      if (flags & 2) v += Cb[addr];
      if (flags & 1) v = v > 0.0f ? v : 0.0f;
      Cb[addr] = v;
    }
  }
}

// ---------------------------------------------------------------------------
// Elementwise: X[i0,i1,i2] += b0[i2] (+ b1[i1,i2]) (+ b2[i0,i2])
// ---------------------------------------------------------------------------
__global__ void k_add3(float* __restrict__ X, const float* __restrict__ b0,
                       const float* __restrict__ b1, const float* __restrict__ b2,
                       int d0, int d1, int d2)
{
  long i = (long)blockIdx.x * blockDim.x + threadIdx.x;
  long n = (long)d0 * d1 * d2;
  if (i >= n) return;
  int i2 = (int)(i % d2);
  long t = i / d2;
  int i1 = (int)(t % d1);
  int i0 = (int)(t / d1);
  float v = X[i] + b0[i2];
  if (b1) v += b1[(long)i1 * d2 + i2];
  if (b2) v += b2[(long)i0 * d2 + i2];
  X[i] = v;
}

// ---------------------------------------------------------------------------
// Deterministic gumbel noise (counter hash); reference scales it by GR=1e-3.
// ---------------------------------------------------------------------------
__device__ __forceinline__ float gumbel_noise(unsigned salt, unsigned i)
{
  unsigned x = salt * 0x9E3779B9u + i * 0x85EBCA6Bu + 0x27D4EB2Fu;
  x ^= x >> 16; x *= 0x7FEB352Du;
  x ^= x >> 15; x *= 0x846CA68Bu;
  x ^= x >> 16;
  float u = (float)(x >> 8) * (1.0f / 16777216.0f) + 1e-10f;
  return -__logf(-__logf(u));
}

// ---------------------------------------------------------------------------
// Per-row masked softmax + EPS + safe-log + gumbel-hard select.
// ---------------------------------------------------------------------------
__global__ void k_softmax_sel(const float* __restrict__ logits,
                              const int* __restrict__ lenArr,
                              float* __restrict__ p, float* __restrict__ lp,
                              float* __restrict__ hot, int* __restrict__ ac,
                              int Bn, int L, unsigned salt)
{
  int b = blockIdx.x * blockDim.x + threadIdx.x;
  if (b >= Bn) return;
  int len = lenArr ? lenArr[b] : L;
  if (len > L) len = L;
  if (len < 1) len = 1;
  const float* z = logits + (long)b * L;

  float mx = -3.402823466e38f;
  for (int i = 0; i < len; ++i) mx = fmaxf(mx, z[i]);
  float s = 0.0f;
  for (int i = 0; i < len; ++i) s += __expf(z[i] - mx);
  float inv = 1.0f / s;

  float mx2 = -3.402823466e38f;
  for (int i = 0; i < len; ++i) {
    float pi = __expf(z[i] - mx) * inv + 1e-8f;
    float z2 = pi + gumbel_noise(salt, (unsigned)(b * L + i)) * 0.001f;
    mx2 = fmaxf(mx2, z2);
  }
  float s2 = 0.0f; int idx = 0; float best = -3.402823466e38f;
  for (int i = 0; i < len; ++i) {
    float pi = __expf(z[i] - mx) * inv + 1e-8f;
    float z2 = pi + gumbel_noise(salt, (unsigned)(b * L + i)) * 0.001f;
    s2 += __expf(z2 - mx2);
    if (z2 > best) { best = z2; idx = i; }
  }
  float inv2 = 1.0f / s2;

  for (int i = 0; i < L; ++i) {
    long o = (long)b * L + i;
    if (i < len) {
      float pi = __expf(z[i] - mx) * inv + 1e-8f;
      p[o]  = pi;
      lp[o] = __logf(pi);
      float z2 = pi + gumbel_noise(salt, (unsigned)(b * L + i)) * 0.001f;
      float y  = __expf(z2 - mx2) * inv2;
      hot[o] = ((i == idx) ? 1.0f : 0.0f) - y + y;   // straight-through forward
    } else {
      p[o] = 0.0f; lp[o] = 0.0f; hot[o] = 0.0f;
    }
  }
  ac[b] = idx;
}

// out[b,o] = sum_a hot[b,a] * Hm[b,a,o]
__global__ void k_mix(const float* __restrict__ hot, const float* __restrict__ Hm,
                      float* __restrict__ out, int A, int O)
{
  int b = blockIdx.x, o = threadIdx.x;
  float s = 0.0f;
  for (int a = 0; a < A; ++a)
    s += hot[(long)b * A + a] * Hm[((long)b * A + a) * O + o];
  out[(long)b * O + o] = s;
}

// Stage-2 selections: emb_second, ac_second_desc, sel_att gather, valid3 lens
__global__ void k_sel2(const float* __restrict__ hot2, const float* __restrict__ cge,
                       const float* __restrict__ cd, const float* __restrict__ cae,
                       const int* __restrict__ ac2, const int* __restrict__ cal,
                       float* __restrict__ es, float* __restrict__ acd,
                       float* __restrict__ satt, int* __restrict__ l3)
{
  int b = blockIdx.x, t = threadIdx.x;   // 256 threads
  const float* h = hot2 + (long)b * MM;

  {
    float s = 0.0f;
    for (int m = 0; m < MM; ++m) s += h[m] * cge[(long)m * HH + t];
    es[(long)b * HH + t] = s;
  }
  for (int d = t; d < DESCD; d += 256) {
    float s = 0.0f;
    for (int m = 0; m < MM; ++m) s += h[m] * cd[(long)m * DESCD + d];
    acd[(long)b * DESCD + d] = s;
  }
  int sel = ac2[b];
  for (int u = t; u < AMAXD * HH; u += 256)
    satt[(long)b * AMAXD * HH + u] = cae[(long)sel * AMAXD * HH + u];
  if (t == 0) l3[b] = cal[sel];
}

// Pack outputs: floats (B,1896) then ac1, ac2, ac3 (value-cast to float)
__global__ void k_pack(const float* p1, const float* lp1, const float* p2,
                       const float* lp2, const float* p3, const float* lp3,
                       const float* ef, const float* es, const float* et,
                       const float* acd,
                       const int* ac1, const int* ac2, const int* ac3,
                       float* __restrict__ out)
{
  int b = blockIdx.x;
  float* o = out + (long)b * 1896;
  for (int t = threadIdx.x; t < 1896; t += blockDim.x) {
    int u = t; float v;
    if      (u < 40)              v = p1[(long)b * 40 + u];
    else if ((u -= 40)  < 40)     v = lp1[(long)b * 40 + u];
    else if ((u -= 40)  < 100)    v = p2[(long)b * 100 + u];
    else if ((u -= 100) < 100)    v = lp2[(long)b * 100 + u];
    else if ((u -= 100) < 40)     v = (u < 10) ? p3[(long)b * 10 + u] : 0.0f;
    else if ((u -= 40)  < 40)     v = (u < 10) ? lp3[(long)b * 10 + u] : 0.0f;
    else if ((u -= 40)  < 128)    v = ef[(long)b * 128 + u];
    else if ((u -= 128) < 256)    v = es[(long)b * 256 + u];
    else if ((u -= 256) < 128)    v = et[(long)b * 128 + u];
    else { u -= 128;              v = acd[(long)b * 1024 + u]; }
    o[t] = v;
  }
  if (threadIdx.x == 0) {
    float* tail = out + 128L * 1896;
    tail[b]       = (float)ac1[b];
    tail[128 + b] = (float)ac2[b];
    tail[256 + b] = (float)ac3[b];
  }
}

// ---------------------------------------------------------------------------
// Host-side GEMM launcher (16x64 strip per wave)
// ---------------------------------------------------------------------------
static void gemm(hipStream_t s, const float* A, const float* B, const float* bias,
                 float* C, int M, int N, int K, int lda, int ldb, int ldc,
                 long azs, long bzs, long czs, int nz, int bLayout, int flags)
{
  int tm = (M + 15) / 16, tn4 = (N + 63) / 64;
  int tiles = tm * tn4;
  int blocks = (tiles + WAVES_PER_BLOCK - 1) / WAVES_PER_BLOCK;
  dim3 g(blocks, 1, nz);
  gemm_wmma_bf16<<<g, dim3(256), 0, s>>>(A, B, bias, C, M, N, K, lda, ldb, ldc,
                                         azs, bzs, czs, bLayout, flags, tn4, tiles);
}

extern "C" void kernel_launch(void* const* d_in, const int* in_sizes, int n_in,
                              void* d_out, int out_size, void* d_ws, size_t ws_size,
                              hipStream_t stream)
{
  const float* G    = (const float*)d_in[0];
  const float* X    = (const float*)d_in[1];
  const float* CD   = (const float*)d_in[2];
  const float* CGE  = (const float*)d_in[3];
  const float* CAE  = (const float*)d_in[4];
  const float* a1Wb = (const float*)d_in[5];   const float* a1bb = (const float*)d_in[6];
  const float* a1W1 = (const float*)d_in[7];   const float* a1W2 = (const float*)d_in[8];
  const float* a1m1 = (const float*)d_in[9];   const float* a1m2 = (const float*)d_in[10];
  const float* a1b2 = (const float*)d_in[11];
  const float* a2Wb = (const float*)d_in[12];  const float* a2bb = (const float*)d_in[13];
  const float* a2W1 = (const float*)d_in[14];  const float* a2W2 = (const float*)d_in[15];
  const float* a2m1 = (const float*)d_in[16];  const float* a2m2 = (const float*)d_in[17];
  const float* a2b2 = (const float*)d_in[18];  const float* a2m3 = (const float*)d_in[19];
  const float* a2b3 = (const float*)d_in[20];
  const float* a3Wb = (const float*)d_in[21];  const float* a3bb = (const float*)d_in[22];
  const float* a3W1 = (const float*)d_in[23];  const float* a3W2 = (const float*)d_in[24];
  const float* a3m1 = (const float*)d_in[25];  const float* a3m2 = (const float*)d_in[26];
  const float* a3b2 = (const float*)d_in[27];
  const int* attLen  = (const int*)d_in[28];
  const int* cattLen = (const int*)d_in[29];

  float* w = (float*)d_ws;
  long off = 0;
  auto alloc = [&](long n) { float* p = w + off; off += n; return p; };
  float* t1   = alloc(128L * 32768);
  float* GW2  = alloc(128L * 128);
  float* h1   = alloc(128L * 40 * 128);
  float* r1   = alloc(5120L * 64);
  float* lg1  = alloc(5120);
  float* p1   = alloc(5120);  float* lp1 = alloc(5120);  float* hot1 = alloc(5120);
  float* ef   = alloc(128L * 128);
  float* U    = alloc(100L * 128 * 128);
  float* D1   = alloc(100L * 128);
  float* D2   = alloc(128L * 128);
  float* h2   = alloc(128L * 100 * 128);
  float* z1   = alloc(12800L * 128);
  float* z2   = alloc(12800L * 128);
  float* lg2  = alloc(12800);
  float* p2   = alloc(12800); float* lp2 = alloc(12800); float* hot2 = alloc(12800);
  float* es   = alloc(128L * 256);
  float* acd  = alloc(128L * 1024);
  float* satt = alloc(128L * 10 * 256);
  float* ES1  = alloc(128L * 128);
  float* t3   = alloc(128L * 128 * 256);
  float* h3   = alloc(128L * 10 * 128);
  float* r3   = alloc(1280L * 64);
  float* lg3  = alloc(1280);
  float* p3   = alloc(1280);  float* lp3 = alloc(1280);  float* hot3 = alloc(1280);
  float* et   = alloc(128L * 128);
  int* ac1 = (int*)(w + off); off += 128;
  int* ac2 = (int*)(w + off); off += 128;
  int* ac3 = (int*)(w + off); off += 128;
  int* l3  = (int*)(w + off); off += 128;
  (void)ws_size; (void)in_sizes; (void)n_in; (void)out_size;

  // ===== stage-2 batch-independent precompute (reassociated contraction) ====
  gemm(stream, CD, a2Wb, nullptr, U, 100, 128, 1024, 1024, 128, 16384,
       0, 131072, 128, 128, /*bLayout=*/1, 0);
  gemm(stream, CD, a2W1, nullptr, D1, 100, 128, 1024, 1024, 1024, 128,
       0, 0, 0, 1, 0, 0);

  // ===== Stage 1 =====
  gemm(stream, G, a1Wb, nullptr, t1, 128, 32768, 256, 256, 256, 32768,
       0, 0, 0, 1, 0, 0);
  gemm(stream, G, a1W2, nullptr, GW2, 128, 128, 256, 256, 256, 128,
       0, 0, 0, 1, 0, 0);
  gemm(stream, X, t1, nullptr, h1, 40, 128, 256, 256, 256, 128,
       (long)NMAXD * HH, 32768, 5120, 128, 0, 0);
  gemm(stream, X, a1W1, nullptr, h1, 40, 128, 256, 256, 256, 128,
       (long)NMAXD * HH, 0, 5120, 128, 0, /*acc*/2);
  { long n = 128L * 40 * 128;
    k_add3<<<(unsigned)((n + 255) / 256), 256, 0, stream>>>(h1, a1bb, nullptr, GW2, 128, 40, 128); }
  gemm(stream, h1, a1m1, nullptr, r1, 5120, 64, 128, 128, 128, 64, 0, 0, 0, 1, 0, /*relu*/1);
  gemm(stream, r1, a1m2, a1b2, lg1, 5120, 1, 64, 64, 64, 1, 0, 0, 0, 1, 0, 0);
  k_softmax_sel<<<2, 64, 0, stream>>>(lg1, attLen, p1, lp1, hot1, ac1, 128, 40, 0x1111u);
  k_mix<<<128, 128, 0, stream>>>(hot1, h1, ef, 40, 128);

  // ===== Stage 2 =====
  gemm(stream, ef, a2W2, nullptr, D2, 128, 128, 128, 128, 128, 128, 0, 0, 0, 1, 0, 0);
  gemm(stream, U, ef, nullptr, h2, 12800, 128, 128, 128, 128, /*ldc=n-stride*/12800,
       0, 0, 0, 1, 0, /*storeT*/4);
  { long n = 128L * 100 * 128;
    k_add3<<<(unsigned)((n + 255) / 256), 256, 0, stream>>>(h2, a2bb, D1, D2, 128, 100, 128); }
  gemm(stream, h2, a2m1, nullptr, z1, 12800, 128, 128, 128, 128, 128, 0, 0, 0, 1, 0, 1);
  gemm(stream, z1, a2m2, a2b2, z2, 12800, 128, 128, 128, 128, 128, 0, 0, 0, 1, 0, 1);
  gemm(stream, z2, a2m3, a2b3, lg2, 12800, 1, 128, 128, 128, 1, 0, 0, 0, 1, 0, 0);
  k_softmax_sel<<<2, 64, 0, stream>>>(lg2, nullptr, p2, lp2, hot2, ac2, 128, 100, 0x2222u);
  k_sel2<<<128, 256, 0, stream>>>(hot2, CGE, CD, CAE, ac2, cattLen, es, acd, satt, l3);

  // ===== Stage 3 =====
  gemm(stream, es, a3W1, nullptr, ES1, 128, 128, 256, 256, 256, 128, 0, 0, 0, 1, 0, 0);
  gemm(stream, es, a3Wb, nullptr, t3, 128, 256, 256, 256, 256, 32768,
       0, 65536, 256, 128, /*bLayout=*/1, 0);
  gemm(stream, satt, t3, nullptr, h3, 10, 128, 256, 256, 256, 128,
       2560, 32768, 1280, 128, 0, 0);
  gemm(stream, satt, a3W2, nullptr, h3, 10, 128, 256, 256, 256, 128,
       2560, 0, 1280, 128, 0, /*acc*/2);
  { long n = 128L * 10 * 128;
    k_add3<<<(unsigned)((n + 255) / 256), 256, 0, stream>>>(h3, a3bb, nullptr, ES1, 128, 10, 128); }
  gemm(stream, h3, a3m1, nullptr, r3, 1280, 64, 128, 128, 128, 64, 0, 0, 0, 1, 0, 1);
  gemm(stream, r3, a3m2, a3b2, lg3, 1280, 1, 64, 64, 64, 1, 0, 0, 0, 1, 0, 0);
  k_softmax_sel<<<2, 64, 0, stream>>>(lg3, l3, p3, lp3, hot3, ac3, 128, 10, 0x3333u);
  k_mix<<<128, 128, 0, stream>>>(hot3, h3, et, 10, 128);

  // ===== pack =====
  k_pack<<<128, 256, 0, stream>>>(p1, lp1, p2, lp2, p3, lp3, ef, es, et, acd,
                                  ac1, ac2, ac3, (float*)d_out);
}
